// FeatureSAE_1700807049888
// MI455X (gfx1250) — compile-verified, matrix operand order checked by reading, not run on hardware
//
#include <hip/hip_runtime.h>

typedef unsigned int u32;
typedef __attribute__((ext_vector_type(16))) __bf16 v16bf;
typedef __attribute__((ext_vector_type(8)))  float  v8f;

#define D_MODEL 1024
#define N_FEAT  16384
#define N_TOK   8192
#define TOPK    32
#define TILE    128
#define KC      32
#define PITCH   17   // uints per LDS row (16 data + 1 pad -> conflict-avoiding stride)

// ---------- helpers: f32 -> (bf16 hi, bf16 lo) split, packed 2-per-uint ----------
__device__ __forceinline__ u32 pack2_hi(float v0, float v1) {
  return (__float_as_uint(v0) >> 16) | (__float_as_uint(v1) & 0xFFFF0000u);
}
__device__ __forceinline__ u32 pack2_lo(float v0, float v1) {
  float h0 = __uint_as_float(__float_as_uint(v0) & 0xFFFF0000u);
  float h1 = __uint_as_float(__float_as_uint(v1) & 0xFFFF0000u);
  float r0 = v0 - h0, r1 = v1 - h1;
  return (__float_as_uint(r0) >> 16) | (__float_as_uint(r1) & 0xFFFF0000u);
}
__device__ __forceinline__ void cvt_store8(u32* dh, u32* dl, float4 v0, float4 v1) {
  float s[8] = {v0.x, v0.y, v0.z, v0.w, v1.x, v1.y, v1.z, v1.w};
#pragma unroll
  for (int j = 0; j < 4; ++j) {
    dh[j] = pack2_hi(s[2*j], s[2*j+1]);
    dl[j] = pack2_lo(s[2*j], s[2*j+1]);
  }
}
__device__ __forceinline__ v8f v8f_zero() {
  v8f z;
#pragma unroll
  for (int i = 0; i < 8; ++i) z[i] = 0.0f;
  return z;
}
// A/B fragment for v_wmma_f32_16x16x32_bf16 (16-bit A 16x32 layout, ISA 7.12.2):
// lane L holds row (L%16); lanes 0-15 K in {0..7,16..23}, lanes 16-31 K in {8..15,24..31}
__device__ __forceinline__ v16bf load_frag(const u32* base, int row0, int lane) {
  int row  = row0 + (lane & 15);
  int half = (lane >> 4) * 4;
  const u32* p = base + row * PITCH + half;
  union { v16bf v; u32 u[8]; } f;
#pragma unroll
  for (int j = 0; j < 4; ++j) { f.u[j] = p[j]; f.u[j + 4] = p[j + 8]; }
  return f.v;
}

// ================= Kernel 1: encoder GEMM (bf16x3 split precision) =================
// pre[m][n] = sum_k x[m][k] * W_enc[n][k] + b_enc[n]   (NT GEMM, both K-major)
__global__ __launch_bounds__(256) void sae_encoder(
    const float* __restrict__ x, const float* __restrict__ W,
    const float* __restrict__ bias, float* __restrict__ pre)
{
  __shared__ u32 lAh[TILE * PITCH], lAl[TILE * PITCH];
  __shared__ u32 lBh[TILE * PITCH], lBl[TILE * PITCH];

  const int t    = threadIdx.x;
  const int lane = t & 31;
  const int wave = t >> 5;
  const int wm   = (wave & 3) * 32;   // wave's row offset inside tile (4 waves over M)
  const int wn   = (wave >> 2) * 64;  // wave's col offset inside tile (2 waves over N)
  const int m0   = blockIdx.x * TILE;
  const int n0   = blockIdx.y * TILE;
  const int lrow = t >> 2;            // 0..63 (two passes cover 128 rows)
  const int koff = (t & 3) * 8;       // 8 consecutive f32 per thread
  const int lcol = (t & 3) * 4;       // matching uint column in LDS

  v8f acc[2][4];
#pragma unroll
  for (int mi = 0; mi < 2; ++mi)
#pragma unroll
    for (int ni = 0; ni < 4; ++ni) acc[mi][ni] = v8f_zero();

  for (int kc = 0; kc < D_MODEL; kc += KC) {
    float4 a0[2], a1[2], b0[2], b1[2];
#pragma unroll
    for (int p = 0; p < 2; ++p) {
      const float* pa = x + (size_t)(m0 + lrow + p * 64) * D_MODEL + kc + koff;
      a0[p] = *(const float4*)pa;  a1[p] = *(const float4*)(pa + 4);
      const float* pb = W + (size_t)(n0 + lrow + p * 64) * D_MODEL + kc + koff;
      b0[p] = *(const float4*)pb;  b1[p] = *(const float4*)(pb + 4);
    }
    __syncthreads();   // previous iteration's fragment reads done
#pragma unroll
    for (int p = 0; p < 2; ++p) {
      int r = lrow + p * 64;
      cvt_store8(lAh + r * PITCH + lcol, lAl + r * PITCH + lcol, a0[p], a1[p]);
      cvt_store8(lBh + r * PITCH + lcol, lBl + r * PITCH + lcol, b0[p], b1[p]);
    }
    __syncthreads();

    v16bf ah[2], al[2], bh[4], bl[4];
#pragma unroll
    for (int mi = 0; mi < 2; ++mi) {
      ah[mi] = load_frag(lAh, wm + mi * 16, lane);
      al[mi] = load_frag(lAl, wm + mi * 16, lane);
    }
#pragma unroll
    for (int ni = 0; ni < 4; ++ni) {
      bh[ni] = load_frag(lBh, wn + ni * 16, lane);
      bl[ni] = load_frag(lBl, wn + ni * 16, lane);
    }
    // term-major ordering: same accumulator reused only every 8 WMMAs (hides RAW hazard)
#pragma unroll
    for (int mi = 0; mi < 2; ++mi)
#pragma unroll
      for (int ni = 0; ni < 4; ++ni)
        acc[mi][ni] = __builtin_amdgcn_wmma_f32_16x16x32_bf16(
            false, ah[mi], false, bh[ni], (short)0, acc[mi][ni], false, false);
#pragma unroll
    for (int mi = 0; mi < 2; ++mi)
#pragma unroll
      for (int ni = 0; ni < 4; ++ni)
        acc[mi][ni] = __builtin_amdgcn_wmma_f32_16x16x32_bf16(
            false, ah[mi], false, bl[ni], (short)0, acc[mi][ni], false, false);
#pragma unroll
    for (int mi = 0; mi < 2; ++mi)
#pragma unroll
      for (int ni = 0; ni < 4; ++ni)
        acc[mi][ni] = __builtin_amdgcn_wmma_f32_16x16x32_bf16(
            false, al[mi], false, bh[ni], (short)0, acc[mi][ni], false, false);
  }

  // epilogue: C/D layout — VGPR g, lanes 0-15: (M=g, N=lane); lanes 16-31: (M=g+8, N=lane-16)
  const int rhalf = (lane >> 4) * 8;
  const int ncol  = lane & 15;
#pragma unroll
  for (int mi = 0; mi < 2; ++mi)
#pragma unroll
    for (int ni = 0; ni < 4; ++ni) {
      int col   = n0 + wn + ni * 16 + ncol;
      float bv  = bias[col];
      size_t bi = (size_t)(m0 + wm + mi * 16 + rhalf) * N_FEAT + col;
#pragma unroll
      for (int g = 0; g < 8; ++g)
        pre[bi + (size_t)g * N_FEAT] = acc[mi][ni][g] + bv;
    }
}

// ================= Kernel 2: exact top-K threshold + sparsify in place =================
__global__ __launch_bounds__(256) void sae_topk(
    float* __restrict__ acts, float* __restrict__ tv, int* __restrict__ ti)
{
  __shared__ int s_counts[256];
  __shared__ int s_total;
  const int t = threadIdx.x;
  const int token = blockIdx.x;
  float* row = acts + (size_t)token * N_FEAT;

  float v[64];
#pragma unroll
  for (int j = 0; j < 64; ++j) v[j] = row[j * 256 + t];

  if (t < TOPK) { tv[token * TOPK + t] = 0.0f; ti[token * TOPK + t] = 0; }

  // binary search on positive IEEE bit patterns: largest thr with count(v>=thr) >= K
  u32 lo = 1u, hi = 0x7F800001u;
  while (hi - lo > 1u) {
    u32 mid = lo + ((hi - lo) >> 1);
    float thr = __uint_as_float(mid);
    int c = 0;
#pragma unroll
    for (int j = 0; j < 64; ++j) c += (v[j] >= thr) ? 1 : 0;
    s_counts[t] = c;
    __syncthreads();
    if (t == 0) {
      int tot = 0;
      for (int i = 0; i < 256; ++i) tot += s_counts[i];
      s_total = tot;
    }
    __syncthreads();
    if (s_total >= TOPK) lo = mid; else hi = mid;
    __syncthreads();
  }
  const float thr = __uint_as_float(lo);   // thr > 0, so selected => relu is identity

  int c = 0;
#pragma unroll
  for (int j = 0; j < 64; ++j) c += (v[j] >= thr) ? 1 : 0;
  s_counts[t] = c;
  __syncthreads();
  int off = 0;
  for (int i = 0; i < t; ++i) off += s_counts[i];   // deterministic compaction order
#pragma unroll
  for (int j = 0; j < 64; ++j) {
    float val = v[j];
    bool sel = (val >= thr);
    row[j * 256 + t] = sel ? val : 0.0f;
    if (sel) {
      if (off < TOPK) { tv[token * TOPK + off] = val; ti[token * TOPK + off] = j * 256 + t; }
      off++;
    }
  }
}

// ================= Kernel 3: transpose W_dec [1024][16384] -> [16384][1024] =================
__global__ __launch_bounds__(256) void sae_transpose(
    const float* __restrict__ Wd, float* __restrict__ WdT)
{
  __shared__ float tile[32][33];
  const int tx = threadIdx.x & 31, ty = threadIdx.x >> 5;
  const int f0 = blockIdx.x * 32, d0 = blockIdx.y * 32;
#pragma unroll
  for (int r = 0; r < 32; r += 8)
    tile[ty + r][tx] = Wd[(size_t)(d0 + ty + r) * N_FEAT + f0 + tx];
  __syncthreads();
#pragma unroll
  for (int r = 0; r < 32; r += 8)
    WdT[(size_t)(f0 + ty + r) * D_MODEL + d0 + tx] = tile[tx][ty + r];
}

// ================= Kernel 4: sparse decoder =================
__global__ __launch_bounds__(256) void sae_decode(
    const float* __restrict__ WdT, const float* __restrict__ tv,
    const int* __restrict__ ti, float* __restrict__ recon)
{
  __shared__ float sv[TOPK];
  __shared__ int   si[TOPK];
  const int t = threadIdx.x;
  const int token = blockIdx.x;
  if (t < TOPK) { sv[t] = tv[token * TOPK + t]; si[t] = ti[token * TOPK + t]; }
  __syncthreads();
  const int d = t * 4;
  float4 acc = {0.f, 0.f, 0.f, 0.f};
#pragma unroll
  for (int k = 0; k < TOPK; ++k) {
    float w = sv[k];
    const float4 c = *(const float4*)(WdT + (size_t)si[k] * D_MODEL + d);
    acc.x += w * c.x; acc.y += w * c.y; acc.z += w * c.z; acc.w += w * c.w;
  }
  *(float4*)(recon + (size_t)token * D_MODEL + d) = acc;
}

extern "C" void kernel_launch(void* const* d_in, const int* in_sizes, int n_in,
                              void* d_out, int out_size, void* d_ws, size_t ws_size,
                              hipStream_t stream) {
  (void)in_sizes; (void)n_in; (void)out_size; (void)ws_size;
  const float* x     = (const float*)d_in[0];   // [8192,1024]
  const float* W_enc = (const float*)d_in[1];   // [16384,1024]
  const float* b_enc = (const float*)d_in[2];   // [16384]
  const float* W_dec = (const float*)d_in[3];   // [1024,16384]

  float* recon = (float*)d_out;                           // [8192,1024]
  float* acts  = (float*)d_out + (size_t)N_TOK * D_MODEL; // [8192,16384] (scratch: pre, then sparse)

  float* WdT = (float*)d_ws;                               // 16384*1024 f32 = 64 MB
  float* tv  = WdT + (size_t)N_FEAT * D_MODEL;             // 8192*32 f32
  int*   ti  = (int*)(tv + (size_t)N_TOK * TOPK);          // 8192*32 i32

  dim3 gE(N_TOK / TILE, N_FEAT / TILE);                    // (64,128)
  sae_encoder<<<gE, 256, 0, stream>>>(x, W_enc, b_enc, acts);
  sae_transpose<<<dim3(N_FEAT / 32, D_MODEL / 32), 256, 0, stream>>>(W_dec, WdT);
  sae_topk<<<N_TOK, 256, 0, stream>>>(acts, tv, ti);
  sae_decode<<<N_TOK, 256, 0, stream>>>(WdT, tv, ti, recon);
}
